// EdgeDecoder_65085934404104
// MI455X (gfx1250) — matrix operand order repaired and checked by reading, compiled, not run on hardware
//
#include <hip/hip_runtime.h>

#define HDIM 128
#define E_TOTAL 1000000
#define NPAIRS (E_TOTAL / 32)   // each wave processes 32 edges (two 16-row M-tiles)

#define LDW1 264
#define LDW2 136
#define LDW3 264
#define LDW4 72
#define LDA  264
#define WAVES 8

typedef __attribute__((ext_vector_type(16))) __bf16 v16bf;
typedef __attribute__((ext_vector_type(8)))  __bf16 v8bf;
typedef __attribute__((ext_vector_type(2)))  __bf16 v2bf;
typedef __attribute__((ext_vector_type(8)))  float  v8f;

#if defined(__has_builtin)
#if __has_builtin(__builtin_amdgcn_cvt_pk_bf16_f32)
#define HAVE_PK_BF16 1
#endif
#endif

__device__ __forceinline__ unsigned short bfbits(float f) {
  union { float f; unsigned u; } v; v.f = f;
  unsigned r = v.u + 0x7FFFu + ((v.u >> 16) & 1u);  // round-to-nearest-even
  return (unsigned short)(r >> 16);
}

// Pack two floats into packed-bf16 dword (hardware v_cvt_pk_bf16_f32 when available).
__device__ __forceinline__ unsigned pack2bf(float a, float b) {
#ifdef HAVE_PK_BF16
  v2bf r = __builtin_amdgcn_cvt_pk_bf16_f32(a, b);
  unsigned u; __builtin_memcpy(&u, &r, 4); return u;
#else
  return (unsigned)bfbits(a) | ((unsigned)bfbits(b) << 16);
#endif
}

__device__ __forceinline__ __bf16 f2bf(float f) {
#ifdef HAVE_PK_BF16
  v2bf r = __builtin_amdgcn_cvt_pk_bf16_f32(f, f);
  return r[0];
#else
  unsigned short b = bfbits(f);
  __bf16 h; __builtin_memcpy(&h, &b, 2); return h;
#endif
}

__device__ __forceinline__ void wavefence() {
  __builtin_amdgcn_fence(__ATOMIC_ACQ_REL, "wavefront");
  __builtin_amdgcn_wave_barrier();
}

// A-fragment (16x32 bf16, M x K): lane = row (l&15); f[0..7]=K[b..b+7],
// f[8..15]=K[b+16..b+23], b=8*(l>>4). Two aligned 16B LDS loads.
__device__ __forceinline__ v16bf load_fragA(const __bf16* base, int ld, int kbase, int lane) {
  const __bf16* p = base + (lane & 15) * ld + kbase + ((lane >> 4) << 3);
  v8bf lo = *(const v8bf*)(p);
  v8bf hi = *(const v8bf*)(p + 16);
  v16bf f;
#pragma unroll
  for (int i = 0; i < 8; ++i) { f[i] = lo[i]; f[i + 8] = hi[i]; }
  return f;
}

// B-fragment (32x16 bf16, K x N) from row-major W[out][in]: lane = out-col (l&15),
// f[0..7]=W[n][b..b+7], f[8..15]=W[n][b+16..b+23], b=kbase+8*(l>>4).
__device__ __forceinline__ v16bf load_fragB(const __bf16* base, int ld, int kbase, int lane) {
  const __bf16* p = base + (lane & 15) * ld + kbase + ((lane >> 4) << 3);
  v8bf lo = *(const v8bf*)(p);
  v8bf hi = *(const v8bf*)(p + 16);
  v16bf f;
#pragma unroll
  for (int i = 0; i < 8; ++i) { f[i] = lo[i]; f[i + 8] = hi[i]; }
  return f;
}

// Writeback of a D-tile (16x16 f32) with fused bias (+ReLU) as bf16 into the
// row-major activation buffer. Bias is per-column == per-lane scalar.
__device__ __forceinline__ void store_tile_bias(__bf16* act, int ld, int n, v8f c,
                                                float bias, int lane, bool relu) {
  const int col = n * 16 + (lane & 15);
  const int r0  = (lane >> 4) << 3;
#pragma unroll
  for (int v = 0; v < 8; ++v) {
    float x = c[v] + bias;
    if (relu) x = fmaxf(x, 0.0f);
    act[(r0 + v) * ld + col] = f2bf(x);
  }
}

#define WMMA_BF16(A, B, C) \
  __builtin_amdgcn_wmma_f32_16x16x32_bf16(false, (A), false, (B), (short)0, (C), false, false)

__global__ __launch_bounds__(256)
void edge_mlp_kernel(const float* __restrict__ z_user,
                     const float* __restrict__ z_book,
                     const int* __restrict__ eidx,   // [2, E]
                     const float* __restrict__ W1, const float* __restrict__ b1,
                     const float* __restrict__ W2, const float* __restrict__ b2,
                     const float* __restrict__ W3, const float* __restrict__ b3,
                     const float* __restrict__ W4, const float* __restrict__ b4,
                     float* __restrict__ out) {
  // ~310 KB static LDS: all weights (bf16, padded rows) + per-wave 32-row activation tiles.
  __shared__ __align__(16) __bf16 sW1[128 * LDW1];
  __shared__ __align__(16) __bf16 sW2[256 * LDW2];
  __shared__ __align__(16) __bf16 sW3[64 * LDW3];
  __shared__ __align__(16) __bf16 sW4[16 * LDW4];
  __shared__ __align__(16) __bf16 sAct[WAVES * 32 * LDA];
  __shared__ float sB1[128], sB2[256], sB3[64], sB4[16];

  const int tid  = threadIdx.x;
  const int lane = tid & 31;
  const int wave = tid >> 5;

  // ---- One-time weight staging (f32 -> bf16, padded row-major) ----
  for (int i = tid; i < 128 * 256; i += 256) sW1[(i >> 8) * LDW1 + (i & 255)] = f2bf(W1[i]);
  for (int i = tid; i < 256 * 128; i += 256) sW2[(i >> 7) * LDW2 + (i & 127)] = f2bf(W2[i]);
  for (int i = tid; i < 64 * 256;  i += 256) sW3[(i >> 8) * LDW3 + (i & 255)] = f2bf(W3[i]);
  for (int i = tid; i < 16 * 64;   i += 256) {
    int r = i >> 6, c = i & 63;
    sW4[r * LDW4 + c] = f2bf(r < 5 ? W4[r * 64 + c] : 0.0f);  // pad outs 5..15 with zero
  }
  for (int i = tid; i < 128; i += 256) sB1[i] = b1[i];
  for (int i = tid; i < 256; i += 256) sB2[i] = b2[i];
  for (int i = tid; i < 64;  i += 256) sB3[i] = b3[i];
  for (int i = tid; i < 16;  i += 256) sB4[i] = (i < 5) ? b4[i] : 0.0f;
  __syncthreads();

  __bf16* act0 = sAct + wave * 32 * LDA;   // M-tile 0: rows 0..15
  __bf16* act1 = act0 + 16 * LDA;          // M-tile 1: rows 16..31

  // Grid-stride over 32-edge pair-tiles; one pair per wave per iteration.
  for (int p = blockIdx.x * WAVES + wave; p < NPAIRS; p += gridDim.x * WAVES) {
    const int e0 = p * 32;

    // ---- Gather: 32 edges x 256 f32 -> bf16 activation rows (1 lane per edge) ----
    {
      const long long iu = (long long)eidx[e0 + lane];
      const long long ib = (long long)eidx[E_TOTAL + e0 + lane];
      const float4* pu = (const float4*)(z_user + iu * HDIM);
      const float4* pb = (const float4*)(z_book + ib * HDIM);
      uint4* du = (uint4*)(act0 + lane * LDA);         // cols 0..127
      uint4* db = (uint4*)(act0 + lane * LDA + 128);   // cols 128..255
#pragma unroll
      for (int j = 0; j < 16; ++j) {
        float4 u0 = pu[2 * j], u1 = pu[2 * j + 1];
        du[j] = make_uint4(pack2bf(u0.x, u0.y), pack2bf(u0.z, u0.w),
                           pack2bf(u1.x, u1.y), pack2bf(u1.z, u1.w));
        float4 v0 = pb[2 * j], v1 = pb[2 * j + 1];
        db[j] = make_uint4(pack2bf(v0.x, v0.y), pack2bf(v0.z, v0.w),
                           pack2bf(v1.x, v1.y), pack2bf(v1.z, v1.w));
      }
    }
    wavefence();

    // ---- Layer 1: 256 -> 128, ReLU (8K x 8N x 2M = 128 WMMA, B loaded once) ----
    {
      v16bf a0[8], a1[8];
#pragma unroll
      for (int k = 0; k < 8; ++k) {
        a0[k] = load_fragA(act0, LDA, k * 32, lane);
        a1[k] = load_fragA(act1, LDA, k * 32, lane);
      }
      wavefence();
#pragma unroll
      for (int n = 0; n < 8; ++n) {
        v8f c0 = {};  // zero C; bias fused at writeback
        v8f c1 = {};
#pragma unroll
        for (int k = 0; k < 8; ++k) {
          v16bf bfr = load_fragB(sW1 + n * 16 * LDW1, LDW1, k * 32, lane);
          c0 = WMMA_BF16(a0[k], bfr, c0);
          c1 = WMMA_BF16(a1[k], bfr, c1);
        }
        const float bias = sB1[n * 16 + (lane & 15)];
        store_tile_bias(act0, LDA, n, c0, bias, lane, true);
        store_tile_bias(act1, LDA, n, c1, bias, lane, true);
      }
    }
    wavefence();

    // ---- Layer 2: 128 -> 256, ReLU (4K x 16N x 2M = 128 WMMA) ----
    {
      v16bf a0[4], a1[4];
#pragma unroll
      for (int k = 0; k < 4; ++k) {
        a0[k] = load_fragA(act0, LDA, k * 32, lane);
        a1[k] = load_fragA(act1, LDA, k * 32, lane);
      }
      wavefence();
#pragma unroll
      for (int n = 0; n < 16; ++n) {
        v8f c0 = {};
        v8f c1 = {};
#pragma unroll
        for (int k = 0; k < 4; ++k) {
          v16bf bfr = load_fragB(sW2 + n * 16 * LDW2, LDW2, k * 32, lane);
          c0 = WMMA_BF16(a0[k], bfr, c0);
          c1 = WMMA_BF16(a1[k], bfr, c1);
        }
        const float bias = sB2[n * 16 + (lane & 15)];
        store_tile_bias(act0, LDA, n, c0, bias, lane, true);
        store_tile_bias(act1, LDA, n, c1, bias, lane, true);
      }
    }
    wavefence();

    // ---- Layer 3: 256 -> 64, ReLU (8K x 4N x 2M = 64 WMMA) ----
    {
      v16bf a0[8], a1[8];
#pragma unroll
      for (int k = 0; k < 8; ++k) {
        a0[k] = load_fragA(act0, LDA, k * 32, lane);
        a1[k] = load_fragA(act1, LDA, k * 32, lane);
      }
      wavefence();
#pragma unroll
      for (int n = 0; n < 4; ++n) {
        v8f c0 = {};
        v8f c1 = {};
#pragma unroll
        for (int k = 0; k < 8; ++k) {
          v16bf bfr = load_fragB(sW3 + n * 16 * LDW3, LDW3, k * 32, lane);
          c0 = WMMA_BF16(a0[k], bfr, c0);
          c1 = WMMA_BF16(a1[k], bfr, c1);
        }
        const float bias = sB3[n * 16 + (lane & 15)];
        store_tile_bias(act0, LDA, n, c0, bias, lane, true);
        store_tile_bias(act1, LDA, n, c1, bias, lane, true);
      }
    }
    wavefence();

    // ---- Layer 4: 64 -> 5 (padded 16), no ReLU (2K x 2M = 4 WMMA), global scatter ----
    {
      v16bf a00 = load_fragA(act0, LDA, 0,  lane);
      v16bf a01 = load_fragA(act0, LDA, 32, lane);
      v16bf a10 = load_fragA(act1, LDA, 0,  lane);
      v16bf a11 = load_fragA(act1, LDA, 32, lane);
      wavefence();
      v8f c0 = {};
      v8f c1 = {};
      v16bf w4f0 = load_fragB(sW4, LDW4, 0,  lane);
      c0 = WMMA_BF16(a00, w4f0, c0);
      c1 = WMMA_BF16(a10, w4f0, c1);
      v16bf w4f1 = load_fragB(sW4, LDW4, 32, lane);
      c0 = WMMA_BF16(a01, w4f1, c0);
      c1 = WMMA_BF16(a11, w4f1, c1);

      const float bias = sB4[lane & 15];
      const int col = lane & 15;
      const int r0  = (lane >> 4) << 3;
      if (col < 5) {
#pragma unroll
        for (int v = 0; v < 8; ++v) {
          out[(long long)(e0 + r0 + v) * 5 + col]      = c0[v] + bias;
          out[(long long)(e0 + 16 + r0 + v) * 5 + col] = c1[v] + bias;
        }
      }
    }
    wavefence();
  }
}

extern "C" void kernel_launch(void* const* d_in, const int* in_sizes, int n_in,
                              void* d_out, int out_size, void* d_ws, size_t ws_size,
                              hipStream_t stream) {
  const float* z_user = (const float*)d_in[0];
  const float* z_book = (const float*)d_in[1];
  const int*   eidx   = (const int*)d_in[2];
  const float* W1 = (const float*)d_in[3];
  const float* b1 = (const float*)d_in[4];
  const float* W2 = (const float*)d_in[5];
  const float* b2 = (const float*)d_in[6];
  const float* W3 = (const float*)d_in[7];
  const float* b3 = (const float*)d_in[8];
  const float* W4 = (const float*)d_in[9];
  const float* b4 = (const float*)d_in[10];
  float* out = (float*)d_out;

  dim3 grid(1024), block(256);
  hipLaunchKernelGGL(edge_mlp_kernel, grid, block, 0, stream,
                     z_user, z_book, eidx, W1, b1, W2, b2, W3, b3, W4, b4, out);
}
